// RPN_17317308137912
// MI455X (gfx1250) — matrix operand order repaired and checked by reading, not compiled, not run on hardware
//
#include <hip/hip_runtime.h>
#include <hip/hip_bf16.h>
#include <math.h>

// ---------------------------------------------------------------------------
// Types for WMMA fragments (CDNA5 / gfx1250, wave32)
// ---------------------------------------------------------------------------
typedef __attribute__((ext_vector_type(16))) _Float16 v16h;
typedef __attribute__((ext_vector_type(8)))  float    v8f;

union HalfVec32 {           // 32 bytes: one f16 WMMA A or B fragment
  uint4    u[2];
  v16h     v;
  _Float16 h[16];
};

// Problem constants
#define BATCH   8
#define FH      64
#define FW      64
#define CIN     512
#define COUT    512
#define NPOS    (FH * FW)              // 4096 per image
#define TOTPOS  (BATCH * NPOS)         // 32768
#define NANCH   9
#define NSLOTS  (NPOS * NANCH)         // 36864 per image
#define NMAXC   18432                  // compacted-anchor capacity (>= true N=18376)
#define MAXOUT  256
#define PH      66                     // padded feature height/width (64 + halo)

// d_out layout (float elements)
#define ROIS_OFF 0
#define EYE_OFF  (BATCH * MAXOUT * 4)                 // 8192
#define OFF_OFF  (EYE_OFF + TOTPOS * 24)              // 794624
#define ANCH_OFF (OFF_OFF + BATCH * MAXOUT * 4)       // 802816

// ---------------------------------------------------------------------------
// Async global->LDS copy (CDNA5 GLOBAL_LOAD_ASYNC_TO_LDS_B128, ASYNCcnt).
// Param 0: pointer to 4 x i32 vector in global AS; param 1: LDS counterpart.
// Falls back to a synchronous copy if this toolchain lacks the builtin.
// ---------------------------------------------------------------------------
#if defined(__HIP_DEVICE_COMPILE__) && __has_builtin(__builtin_amdgcn_global_load_async_to_lds_b128)
#define HAVE_ASYNC_LDS 1
#else
#define HAVE_ASYNC_LDS 0
#endif

typedef int v4i_t __attribute__((__vector_size__(4 * sizeof(int))));
typedef __attribute__((address_space(1))) v4i_t GV4;
typedef __attribute__((address_space(3))) v4i_t LV4;

__device__ __forceinline__ void copy16_g2l(const _Float16* __restrict__ g,
                                           _Float16* __restrict__ l) {
#if HAVE_ASYNC_LDS
  __builtin_amdgcn_global_load_async_to_lds_b128((GV4*)g, (LV4*)l, 0, 0);
#else
  *(uint4*)l = *(const uint4*)g;
#endif
}

__device__ __forceinline__ void wait_async_lds() {
#if HAVE_ASYNC_LDS
#if __has_builtin(__builtin_amdgcn_s_wait_asynccnt)
  __builtin_amdgcn_s_wait_asynccnt(0);
#else
  asm volatile("s_wait_asynccnt 0x0" ::: "memory");
#endif
#endif
}

// ---------------------------------------------------------------------------
// Anchor geometry (matches _generate_anchors; xs == ys == 16)
// Returns total number of valid (in-image) anchors N.
// ---------------------------------------------------------------------------
__device__ __forceinline__ int anchor_tables(int* kminx, int* kmaxx,
                                             int* kminy, int* kmaxy,
                                             float* wa, float* ha) {
  const float scales[3] = {8.f, 16.f, 32.f};
  const float ratios[3] = {0.5f, 1.f, 2.f};
  int N = 0;
#pragma unroll
  for (int t = 0; t < 9; ++t) {
    float S = scales[t % 3];
    float R = ratios[t / 3];
    float h = sqrtf(S * S / R) * 16.0f;
    float w = h * R;                 // * xs / ys == 1
    wa[t] = w; ha[t] = h;
    float hw = 0.5f * w, hh = 0.5f * h;
    int kx0 = (int)ceilf((hw - 8.0f) * (1.0f / 16.0f));     if (kx0 < 0)  kx0 = 0;
    int kx1 = (int)floorf((1016.0f - hw) * (1.0f / 16.0f)); if (kx1 > 63) kx1 = 63;
    int ky0 = (int)ceilf((hh - 8.0f) * (1.0f / 16.0f));     if (ky0 < 0)  ky0 = 0;
    int ky1 = (int)floorf((1016.0f - hh) * (1.0f / 16.0f)); if (ky1 > 63) ky1 = 63;
    kminx[t] = kx0; kmaxx[t] = kx1; kminy[t] = ky0; kmaxy[t] = ky1;
    int cx = kx1 - kx0 + 1; if (cx < 0) cx = 0;
    int cy = ky1 - ky0 + 1; if (cy < 0) cy = 0;
    N += cx * cy;
  }
  return N;
}

// ---------------------------------------------------------------------------
// prep_f: features (B,64,64,512) f32 -> zero-padded f16 image (B,66,66,512)
// so conv staging is a pure, branch-free byte copy (async-DMA-able).
// ---------------------------------------------------------------------------
__global__ void prep_f_kernel(const float* __restrict__ feat, _Float16* __restrict__ featP) {
  int i = blockIdx.x * 256 + threadIdx.x;           // over B*66*66*512
  if (i >= BATCH * PH * PH * CIN) return;
  int c  = i & (CIN - 1);
  int xx = (i >> 9) % PH;
  int r  = (i >> 9) / PH;
  int yy = r % PH;
  int b  = r / PH;
  float v = 0.f;
  if (xx >= 1 && xx <= FW && yy >= 1 && yy <= FH)
    v = feat[(((b * FH) + (yy - 1)) * FW + (xx - 1)) * CIN + c];
  featP[i] = (_Float16)v;
}

// ---------------------------------------------------------------------------
// prep_w: conv1_w (3,3,IC,OC) f32  ->  wt[tap][oc][ic] f16 (transposed)
// ---------------------------------------------------------------------------
__global__ void prep_w_kernel(const float* __restrict__ w, _Float16* __restrict__ wt) {
  int i = blockIdx.x * 256 + threadIdx.x;           // over 9*512*512
  if (i >= 9 * COUT * CIN) return;
  int ic = i & (CIN - 1);
  int oc = (i >> 9) & (COUT - 1);
  int t  = i >> 18;
  wt[i] = (_Float16)w[(t * CIN + ic) * COUT + oc];
}

// ---------------------------------------------------------------------------
// prep_cat: build WcatT[80][512] f16 (reg|cls|eye, zero-padded) + biasCat[80]
// ---------------------------------------------------------------------------
__global__ void prep_cat_kernel(const float* __restrict__ reg_w, const float* __restrict__ cls_w,
                                const float* __restrict__ eye_w, const float* __restrict__ reg_b,
                                const float* __restrict__ cls_b, const float* __restrict__ eye_b,
                                _Float16* __restrict__ wcatT, float* __restrict__ biasCat) {
  int i = blockIdx.x * 256 + threadIdx.x;           // over 80*512
  if (i >= 80 * CIN) return;
  int k = i & (CIN - 1);
  int n = i >> 9;
  float v = 0.f;
  if (n < 36)      v = reg_w[k * 36 + n];
  else if (n < 45) v = cls_w[k * 9 + (n - 36)];
  else if (n < 69) v = eye_w[k * 24 + (n - 45)];
  wcatT[n * CIN + k] = (_Float16)v;
  if (k == 0) {
    float bvv = 0.f;
    if (n < 36)      bvv = reg_b[n];
    else if (n < 45) bvv = cls_b[n - 36];
    else if (n < 69) bvv = eye_b[n - 45];
    biasCat[n] = bvv;
  }
}

// ---------------------------------------------------------------------------
// conv3x3 + bias + relu, implicit GEMM via v_wmma_f32_16x16x32_f16.
// Workgroup tile: M=64 (one image row) x N=128; 8 waves = 4(M) x 2(N),
// each wave computes 16x64 (four 16x16 accumulators).
// K loop: 9 taps x 8 chunks of 64 input channels = 72 chunks, double-buffered
// LDS with async global->LDS staging overlapped against WMMA compute.
// Fragment loads are grouped (A + all 4 B) so the 4 WMMAs issue back-to-back
// behind a single s_wait_dscnt.
// ---------------------------------------------------------------------------
__global__ __launch_bounds__(256) void conv3x3_kernel(const _Float16* __restrict__ featP,
                                                      const _Float16* __restrict__ wt,
                                                      const float* __restrict__ bias,
                                                      _Float16* __restrict__ xH) {
  const int y0    = blockIdx.x;           // image row 0..63
  const int nBase = blockIdx.y * 128;     // output-channel tile base
  const int b     = blockIdx.z;
  const int tid   = threadIdx.x;
  const int lane  = tid & 31;
  const int wave  = tid >> 5;
  const int waveM = wave & 3;             // 4 waves along M
  const int waveN = wave >> 2;            // 2 waves along N
  const int lhalf = (lane >= 16) ? 1 : 0;

  __shared__ _Float16 Alds[2][64 * 64];   // [m][k]   8 KB per buffer
  __shared__ _Float16 Blds[2][128 * 64];  // [n][k]  16 KB per buffer

  v8f acc[4];
#pragma unroll
  for (int i = 0; i < 4; ++i)
#pragma unroll
    for (int j = 0; j < 8; ++j) acc[i][j] = 0.0f;

  const int mRow = waveM * 16 + (lane & 15);
  const int kbA  = lhalf ? 8 : 0;         // A fragment K base (per ISA layout)
  const int kloB = lhalf ? 8 : 0;         // B fragment K base (per ISA layout)
  const int nCol = waveN * 64 + (lane & 15);

  // Staging decomposition (per thread): A 2x16B, B 4x16B
  const int aRow = tid >> 2;              // 0..63 (x position)
  const int aOff = (tid & 3) * 16;        // halves: 0,16,32,48
  const int bRow = tid >> 1;              // 0..127 (oc within tile)
  const int bOff = (tid & 1) * 32;        // halves: 0,32

  // stage chunk q (q = tap*8 + icChunk) into buffer `buf`
  auto stage = [&](int buf, int q) {
    const int t      = q >> 3;
    const int icBase = (q & 7) * 64;
    const int ky = t / 3, kx = t % 3;     // padded image: no -1 offsets
    const int yy = y0 + ky;
    const _Float16* asrc =
        featP + (((b * PH + yy) * PH + (aRow + kx)) * CIN + icBase + aOff);
    _Float16* adst = &Alds[buf][aRow * 64 + aOff];
    copy16_g2l(asrc, adst);
    copy16_g2l(asrc + 8, adst + 8);
    const _Float16* bsrc =
        wt + ((t * COUT + (nBase + bRow)) * CIN + icBase + bOff);
    _Float16* bdst = &Blds[buf][bRow * 64 + bOff];
    copy16_g2l(bsrc,      bdst);
    copy16_g2l(bsrc + 8,  bdst + 8);
    copy16_g2l(bsrc + 16, bdst + 16);
    copy16_g2l(bsrc + 24, bdst + 24);
  };

  stage(0, 0);
  wait_async_lds();
  __syncthreads();

#pragma unroll 1
  for (int q = 0; q < 72; ++q) {
    if (q + 1 < 72) stage((q + 1) & 1, q + 1);   // overlap DMA with compute
    const _Float16* A  = Alds[q & 1];
    const _Float16* Bl = Blds[q & 1];
#pragma unroll
    for (int s = 0; s < 2; ++s) {                // two K=32 steps per chunk
      // ---- load A fragment and ALL four B fragments first ...
      HalfVec32 af;
      af.u[0] = *(const uint4*)&A[mRow * 64 + s * 32 + kbA];
      af.u[1] = *(const uint4*)&A[mRow * 64 + s * 32 + kbA + 16];
      HalfVec32 bf[4];
#pragma unroll
      for (int nt = 0; nt < 4; ++nt) {
        const int n = nCol + nt * 16;
        bf[nt].u[0] = *(const uint4*)&Bl[n * 64 + s * 32 + kloB];
        bf[nt].u[1] = *(const uint4*)&Bl[n * 64 + s * 32 + kloB + 16];
      }
      // ---- ... then issue the 4 independent WMMAs back-to-back
#pragma unroll
      for (int nt = 0; nt < 4; ++nt) {
        acc[nt] = __builtin_amdgcn_wmma_f32_16x16x32_f16(
            false, af.v, false, bf[nt].v, (short)0, acc[nt], false, false);
      }
    }
    wait_async_lds();
    __syncthreads();
  }

  // ---- epilogue: bias + relu, store x as f16
#pragma unroll
  for (int nt = 0; nt < 4; ++nt) {
    const int n = nBase + waveN * 64 + nt * 16 + (lane & 15);
    const float bv = bias[n];
#pragma unroll
    for (int r = 0; r < 8; ++r) {
      const int mx = waveM * 16 + r + (lhalf ? 8 : 0);
      float v = acc[nt][r] + bv;
      v = v > 0.f ? v : 0.f;
      xH[((b * FH + y0) * FW + mx) * COUT + n] = (_Float16)v;
    }
  }
}

// ---------------------------------------------------------------------------
// proj: x[16 rows x 512] @ WcatT^T[512 x 80]  -> deltas(36), sigmoid scores(9),
// eye(24, written straight to d_out). One wave per 16-row block.
// ---------------------------------------------------------------------------
__global__ __launch_bounds__(256) void proj_kernel(const _Float16* __restrict__ xH,
                                                   const _Float16* __restrict__ wcatT,
                                                   const float* __restrict__ biasCat,
                                                   float* __restrict__ deltas_raw,
                                                   float* __restrict__ scores_raw,
                                                   float* __restrict__ out_eye) {
  const int lane  = threadIdx.x & 31;
  const int wave  = threadIdx.x >> 5;
  const int blk   = blockIdx.x * 8 + wave;      // 0..2047
  const int lhalf = (lane >= 16) ? 1 : 0;
  const int m     = lane & 15;
  const int posA  = blk * 16 + m;
  const int kbA   = lhalf ? 8 : 0;
  const int kloB  = lhalf ? 8 : 0;

  v8f acc[5];
#pragma unroll
  for (int i = 0; i < 5; ++i)
#pragma unroll
    for (int j = 0; j < 8; ++j) acc[i][j] = 0.0f;

#pragma unroll 1
  for (int kc = 0; kc < 16; ++kc) {
    const int k0 = kc * 32;
    HalfVec32 afrag;
    afrag.u[0] = *(const uint4*)&xH[posA * CIN + k0 + kbA];
    afrag.u[1] = *(const uint4*)&xH[posA * CIN + k0 + kbA + 16];
    HalfVec32 bf[5];
#pragma unroll
    for (int nt = 0; nt < 5; ++nt) {
      const int n = nt * 16 + m;
      bf[nt].u[0] = *(const uint4*)&wcatT[n * CIN + k0 + kloB];
      bf[nt].u[1] = *(const uint4*)&wcatT[n * CIN + k0 + kloB + 16];
    }
#pragma unroll
    for (int nt = 0; nt < 5; ++nt) {
      acc[nt] = __builtin_amdgcn_wmma_f32_16x16x32_f16(
          false, afrag.v, false, bf[nt].v, (short)0, acc[nt], false, false);
    }
  }

#pragma unroll
  for (int nt = 0; nt < 5; ++nt) {
    const int n = nt * 16 + m;
    const float bv = (n < 80) ? biasCat[n] : 0.f;
#pragma unroll
    for (int r = 0; r < 8; ++r) {
      const int mrow = r + (lhalf ? 8 : 0);
      const int pos  = blk * 16 + mrow;
      const float v  = acc[nt][r] + bv;
      if (n < 36) {
        deltas_raw[pos * 36 + n] = v;
      } else if (n < 45) {
        scores_raw[pos * 9 + (n - 36)] = 1.0f / (1.0f + expf(-v));
      } else if (n < 69) {
        out_eye[pos * 24 + (n - 45)] = v;
      }
    }
  }
}

// ---------------------------------------------------------------------------
// decode: anchors + mask (closed-form compaction) + box decode + compaction.
// ---------------------------------------------------------------------------
__global__ void decode_kernel(const float* __restrict__ deltas_raw,
                              const float* __restrict__ scores_raw,
                              float* __restrict__ boxes_ws,
                              float* __restrict__ scoresC,
                              float* __restrict__ deltasC,
                              float* __restrict__ out_anchors) {
  const int i = blockIdx.x * 256 + threadIdx.x;
  if (i >= NSLOTS) return;
  const int a   = i % 9;
  const int pos = i / 9;
  const int x   = pos & 63;
  const int y   = pos >> 6;

  int kminx[9], kmaxx[9], kminy[9], kmaxy[9];
  float wa[9], ha[9];
  const int N = anchor_tables(kminx, kmaxx, kminy, kmaxy, wa, ha);

  const bool inX = (x >= kminx[a]) && (x <= kmaxx[a]);
  const bool inY = (y >= kminy[a]) && (y <= kmaxy[a]);
  if (!(inX && inY)) return;

  // compact index = #valid anchors strictly before slot i (separable mask)
  int j = 0;
#pragma unroll
  for (int t = 0; t < 9; ++t) {
    int cntx = kmaxx[t] - kminx[t] + 1; if (cntx < 0) cntx = 0;
    int rb = y - 1; if (rb > kmaxy[t]) rb = kmaxy[t];
    int rowsBefore = rb - kminy[t] + 1; if (rowsBefore < 0) rowsBefore = 0;
    j += cntx * rowsBefore;
    const bool inYt = (y >= kminy[t]) && (y <= kmaxy[t]);
    if (inYt) {
      int xb = x - 1; if (xb > kmaxx[t]) xb = kmaxx[t];
      int xBefore = xb - kminx[t] + 1; if (xBefore < 0) xBefore = 0;
      j += xBefore;
      if (t < a) {
        const bool inXt = (x >= kminx[t]) && (x <= kmaxx[t]);
        j += inXt ? 1 : 0;
      }
    }
  }

  const float cx = 8.0f + 16.0f * (float)x;
  const float cy = 8.0f + 16.0f * (float)y;
  const float x1 = cx - wa[a] * 0.5f, x2 = cx + wa[a] * 0.5f;
  const float y1 = cy - ha[a] * 0.5f, y2 = cy + ha[a] * 0.5f;

#pragma unroll 1
  for (int b = 0; b < BATCH; ++b) {
    float* dst = out_anchors + ((long)b * N + j) * 4;
    dst[0] = x1; dst[1] = y1; dst[2] = x2; dst[3] = y2;
  }

  const float acx = (x1 + x2) * 0.5f, acy = (y1 + y2) * 0.5f;
  const float aw  = x2 - x1,          ah  = y2 - y1;

#pragma unroll 1
  for (int b = 0; b < BATCH; ++b) {
    const float* dr = deltas_raw + (long)(b * NPOS + pos) * 36 + a * 4;
    const float d0 = dr[0], d1 = dr[1], d2 = dr[2], d3 = dr[3];
    const float pcx = d0 * aw + acx;
    const float pcy = d1 * ah + acy;
    const float pw  = expf(d2) * aw;
    const float ph  = expf(d3) * ah;
    float bx1 = pcx - pw * 0.5f, by1 = pcy - ph * 0.5f;
    float bx2 = pcx + pw * 0.5f, by2 = pcy + ph * 0.5f;
    bx1 = fminf(fmaxf(bx1, 0.f), 1024.f);
    by1 = fminf(fmaxf(by1, 0.f), 1024.f);
    bx2 = fminf(fmaxf(bx2, 0.f), 1024.f);
    by2 = fminf(fmaxf(by2, 0.f), 1024.f);
    float* bp = boxes_ws + ((long)b * NMAXC + j) * 4;
    bp[0] = bx1; bp[1] = by1; bp[2] = bx2; bp[3] = by2;
    scoresC[b * NMAXC + j] = scores_raw[(long)(b * NPOS + pos) * 9 + a];
    float* dp = deltasC + ((long)b * NMAXC + j) * 4;
    dp[0] = d0; dp[1] = d1; dp[2] = d2; dp[3] = d3;
  }
}

// ---------------------------------------------------------------------------
// nms: greedy NMS per batch, scores live in LDS (CDNA5 320KB/WGP).
// Tie-break identical to jnp.argmax (first max). Self-suppression included.
// ---------------------------------------------------------------------------
__global__ __launch_bounds__(1024) void nms_kernel(const float* __restrict__ boxes_ws,
                                                   const float* __restrict__ scoresC,
                                                   const float* __restrict__ deltasC,
                                                   float* __restrict__ out_rois,
                                                   float* __restrict__ out_offsets) {
  __shared__ float s_sc[NMAXC];
  __shared__ float s_red[1024];
  __shared__ int   s_idx[1024];
  __shared__ int   s_sel[MAXOUT];
  __shared__ float s_box[5];

  const int b   = blockIdx.x;
  const int tid = threadIdx.x;

  int kminx[9], kmaxx[9], kminy[9], kmaxy[9];
  float wa[9], ha[9];
  const int N = anchor_tables(kminx, kmaxx, kminy, kmaxy, wa, ha);

  const float* boxes = boxes_ws + (long)b * NMAXC * 4;
  for (int i = tid; i < N; i += 1024) s_sc[i] = scoresC[b * NMAXC + i];
  __syncthreads();

  const float NEGINF = -__builtin_inff();
#pragma unroll 1
  for (int it = 0; it < MAXOUT; ++it) {
    // local argmax with first-occurrence tie-break
    float bv = NEGINF;
    int   bi = 0x7fffffff;
    for (int i = tid; i < N; i += 1024) {
      const float v = s_sc[i];
      if (v > bv || (v == bv && i < bi)) { bv = v; bi = i; }
    }
    s_red[tid] = bv; s_idx[tid] = bi;
    __syncthreads();
    for (int s = 512; s > 0; s >>= 1) {
      if (tid < s) {
        const float v2 = s_red[tid + s];
        const int   i2 = s_idx[tid + s];
        if (v2 > s_red[tid] || (v2 == s_red[tid] && i2 < s_idx[tid])) {
          s_red[tid] = v2; s_idx[tid] = i2;
        }
      }
      __syncthreads();
    }
    if (tid == 0) {
      const int ii = s_idx[0];
      s_sel[it] = ii;
      const float* bp = boxes + (long)ii * 4;
      s_box[0] = bp[0]; s_box[1] = bp[1]; s_box[2] = bp[2]; s_box[3] = bp[3];
      s_box[4] = (bp[2] - bp[0]) * (bp[3] - bp[1]);
    }
    __syncthreads();
    const float bx1 = s_box[0], by1 = s_box[1], bx2 = s_box[2], by2 = s_box[3];
    const float barea = s_box[4];
    for (int i = tid; i < N; i += 1024) {
      const float* p = boxes + (long)i * 4;
      float iw = fminf(p[2], bx2) - fmaxf(p[0], bx1); iw = fmaxf(iw, 0.f);
      float ih = fminf(p[3], by2) - fmaxf(p[1], by1); ih = fmaxf(ih, 0.f);
      const float inter = iw * ih;
      const float area  = (p[2] - p[0]) * (p[3] - p[1]);
      const float iou   = inter / (area + barea - inter);
      if (iou > 0.5f) s_sc[i] = NEGINF;
    }
    __syncthreads();
  }

  if (tid < MAXOUT) {
    const int ii = s_sel[tid];
    const float* p = boxes + (long)ii * 4;
    float* r = out_rois + (b * MAXOUT + tid) * 4;
    r[0] = (p[0] + p[2]) * 0.5f;
    r[1] = (p[1] + p[3]) * 0.5f;
    r[2] = p[2] - p[0];
    r[3] = p[3] - p[1];
    const float* dc = deltasC + ((long)b * NMAXC + ii) * 4;
    float* o = out_offsets + (b * MAXOUT + tid) * 4;
    o[0] = dc[0]; o[1] = dc[1]; o[2] = dc[2]; o[3] = dc[3];
  }
}

// ---------------------------------------------------------------------------
// kernel_launch
// ---------------------------------------------------------------------------
extern "C" void kernel_launch(void* const* d_in, const int* in_sizes, int n_in,
                              void* d_out, int out_size, void* d_ws, size_t ws_size,
                              hipStream_t stream) {
  (void)in_sizes; (void)n_in; (void)out_size; (void)ws_size;
  const float* features = (const float*)d_in[0];
  // d_in[1] = image (unused by the reference)
  const float* conv1_w  = (const float*)d_in[2];
  const float* conv1_b  = (const float*)d_in[3];
  const float* reg_w    = (const float*)d_in[4];
  const float* reg_b    = (const float*)d_in[5];
  const float* cls_w    = (const float*)d_in[6];
  const float* cls_b    = (const float*)d_in[7];
  const float* eye_w    = (const float*)d_in[8];
  const float* eye_b    = (const float*)d_in[9];
  float* out = (float*)d_out;

  char* ws = (char*)d_ws;
  size_t o = 0;
  auto carve = [&](size_t bytes) -> char* {
    char* p = ws + o;
    o = (o + bytes + 255) & ~(size_t)255;
    return p;
  };
  _Float16* featP    = (_Float16*)carve((size_t)BATCH * PH * PH * CIN * 2); // 35.7 MB
  _Float16* wt       = (_Float16*)carve((size_t)9 * COUT * CIN * 2);        // 4.7 MB
  _Float16* xH       = (_Float16*)carve((size_t)TOTPOS * CIN * 2);          // 33.5 MB
  _Float16* wcatT    = (_Float16*)carve((size_t)80 * CIN * 2);
  float*    biasCat  = (float*)carve(80 * 4);
  float*    deltasR  = (float*)carve((size_t)TOTPOS * 36 * 4);              // 4.7 MB
  float*    scoresR  = (float*)carve((size_t)TOTPOS * 9 * 4);               // 1.2 MB
  float*    boxesC   = (float*)carve((size_t)BATCH * NMAXC * 4 * 4);        // 2.4 MB
  float*    scoresC  = (float*)carve((size_t)BATCH * NMAXC * 4);
  float*    deltasC  = (float*)carve((size_t)BATCH * NMAXC * 4 * 4);

  // 1) input/weight prep
  prep_f_kernel<<<(BATCH * PH * PH * CIN + 255) / 256, 256, 0, stream>>>(features, featP);
  prep_w_kernel<<<(9 * COUT * CIN + 255) / 256, 256, 0, stream>>>(conv1_w, wt);
  prep_cat_kernel<<<(80 * CIN + 255) / 256, 256, 0, stream>>>(
      reg_w, cls_w, eye_w, reg_b, cls_b, eye_b, wcatT, biasCat);

  // 2) conv 3x3 + relu (implicit GEMM, WMMA, async LDS double-buffering)
  conv3x3_kernel<<<dim3(FH, COUT / 128, BATCH), 256, 0, stream>>>(
      featP, wt, conv1_b, xH);

  // 3) heads (WMMA): deltas / sigmoid-scores / eye
  proj_kernel<<<TOTPOS / (16 * 8), 256, 0, stream>>>(
      xH, wcatT, biasCat, deltasR, scoresR, out + EYE_OFF);

  // 4) anchors + decode + compaction
  decode_kernel<<<(NSLOTS + 255) / 256, 256, 0, stream>>>(
      deltasR, scoresR, boxesC, scoresC, deltasC, out + ANCH_OFF);

  // 5) per-batch NMS + gather rois/offsets
  nms_kernel<<<BATCH, 1024, 0, stream>>>(
      boxesC, scoresC, deltasC, out + ROIS_OFF, out + OFF_OFF);
}
